// Model_53841710023370
// MI455X (gfx1250) — matrix operand order repaired and checked by reading, not compiled, not run on hardware
//
#include <hip/hip_runtime.h>
#include <hip/hip_bf16.h>

// ---------------- problem constants (from reference) ----------------
#define NN 20000      // nodes
#define EE 320000     // edges
#define DIN 256       // input feature dim
#define HIDW 128      // hidden dim
#define NLAYERS 2

typedef __attribute__((ext_vector_type(16))) __bf16 v16bf;
typedef __attribute__((ext_vector_type(8)))  __bf16 v8bf;
typedef __attribute__((ext_vector_type(8)))  float  v8f;

#define WMMA_BF16(a, b, c) \
    __builtin_amdgcn_wmma_f32_16x16x32_bf16(false, (a), false, (b), (short)0, (c), false, false)

// ---------------- small elementwise kernels ----------------
__global__ void fill_zero_kernel(float* p, int n) {
    int i = blockIdx.x * blockDim.x + threadIdx.x;
    if (i < n) p[i] = 0.f;
}

__global__ void deg_count_kernel(const int* __restrict__ dst, float* __restrict__ deg, int e) {
    int i = blockIdx.x * blockDim.x + threadIdx.x;
    if (i < e) atomicAdd(&deg[dst[i]], 1.0f);
}

__global__ void inv_deg_kernel(const float* __restrict__ deg, float* __restrict__ inv, int n) {
    int i = blockIdx.x * blockDim.x + threadIdx.x;
    if (i < n) inv[i] = 1.0f / fmaxf(deg[i], 1.0f);
}

__global__ void cvt_bf16_kernel(const float* __restrict__ in, __bf16* __restrict__ out, int n) {
    int i = blockIdx.x * blockDim.x + threadIdx.x;
    if (i < n) out[i] = (__bf16)in[i];
}

// out_bf16[i] = (agg[i] * inv_deg[i >> shift])   (shift = log2(feature dim))
__global__ void scale_cvt_kernel(const float* __restrict__ agg, const float* __restrict__ inv,
                                 __bf16* __restrict__ out, int shift, int n) {
    int i = blockIdx.x * blockDim.x + threadIdx.x;
    if (i < n) out[i] = (__bf16)(agg[i] * inv[i >> shift]);
}

// segment-sum over edges: agg[dst] += f[src]; one thread handles one edge x 4 features
// dshift = log2(D); total = E << (dshift-2)
__global__ void scatter_add_kernel(const float* __restrict__ f, const int* __restrict__ src,
                                   const int* __restrict__ dst, float* __restrict__ agg,
                                   int dshift, int total) {
    int idx = blockIdx.x * blockDim.x + threadIdx.x;
    if (idx >= total) return;
    int cshift = dshift - 2;
    int e = idx >> cshift;
    int c = (idx & ((1 << cshift) - 1)) << 2;
    int s = src[e], d = dst[e];
    const float4 v = *(const float4*)(f + ((size_t)s << dshift) + c);
    float* o = agg + ((size_t)d << dshift) + c;
    atomicAdd(o + 0, v.x);
    atomicAdd(o + 1, v.y);
    atomicAdd(o + 2, v.z);
    atomicAdd(o + 3, v.w);
}

// ---------------- WMMA fragment loaders (ISA 7.12.2 layouts) ----------------
// A (16x32 bf16): lane r=lane&15 is row M; kg=lane>>4; element i -> k = k0+8*kg + (i<8 ? i : 8+i)
__device__ __forceinline__ v16bf load_a_frag(const __bf16* __restrict__ A, int K,
                                             int node_base, int k0, int lane) {
    int r = lane & 15, kg = lane >> 4;
    const __bf16* p = A + (size_t)(node_base + r) * K + k0 + 8 * kg;
    v8bf lo = *(const v8bf*)p;          // k0+8kg + 0..7
    v8bf hi = *(const v8bf*)(p + 16);   // k0+8kg + 16..23
    return __builtin_shufflevector(lo, hi, 0,1,2,3,4,5,6,7,8,9,10,11,12,13,14,15);
}

// B (32x16 bf16) from W stored [128 x K] row-major (out = f @ W^T):
// lane c=lane&15 is column N; kg=lane>>4; element i -> k = k0 + 8*kg + (i<8 ? i : 8+i)
__device__ __forceinline__ v16bf load_b_frag(const __bf16* __restrict__ W, int K,
                                             int col_base, int k0, int lane) {
    int c = lane & 15, kg = lane >> 4;
    const __bf16* p = W + (size_t)(col_base + c) * K + k0 + 8 * kg;
    v8bf lo = *(const v8bf*)p;
    v8bf hi = *(const v8bf*)(p + 16);
    return __builtin_shufflevector(lo, hi, 0,1,2,3,4,5,6,7,8,9,10,11,12,13,14,15);
}

// ---------------- fused z/r gate GEMM ----------------
// z_pre = agg_x@Wlxz^T + x@Wrxz^T + agg_h@Wlhz^T + h@Wrhz^T + bxz + bhz
// r_pre = same with xr/hr weights; emits z and rh = sigmoid(r_pre)*h
__global__ void __launch_bounds__(256)
gate_zr_kernel(const __bf16* __restrict__ aggx, const __bf16* __restrict__ xbf, int Kx,
               const __bf16* __restrict__ agghbf, const __bf16* __restrict__ hbf,
               const __bf16* __restrict__ Wlxz, const __bf16* __restrict__ Wrxz,
               const __bf16* __restrict__ Wlhz, const __bf16* __restrict__ Wrhz,
               const __bf16* __restrict__ Wlxr, const __bf16* __restrict__ Wrxr,
               const __bf16* __restrict__ Wlhr, const __bf16* __restrict__ Wrhr,
               const float* __restrict__ bxz, const float* __restrict__ bhz,
               const float* __restrict__ bxr, const float* __restrict__ bhr,
               const float* __restrict__ hfull,
               float* __restrict__ zout, float* __restrict__ rhout) {
    int lane = threadIdx.x & 31;
    int wave = threadIdx.x >> 5;          // 0..7 -> output column tile
    int node_base = blockIdx.x * 16;
    int col_base = wave * 16;

    v8f accz = {};
    v8f accr = {};
    for (int k0 = 0; k0 < Kx; k0 += 32) {
        v16bf a = load_a_frag(aggx, Kx, node_base, k0, lane);
        accz = WMMA_BF16(a, load_b_frag(Wlxz, Kx, col_base, k0, lane), accz);
        accr = WMMA_BF16(a, load_b_frag(Wlxr, Kx, col_base, k0, lane), accr);
        a = load_a_frag(xbf, Kx, node_base, k0, lane);
        accz = WMMA_BF16(a, load_b_frag(Wrxz, Kx, col_base, k0, lane), accz);
        accr = WMMA_BF16(a, load_b_frag(Wrxr, Kx, col_base, k0, lane), accr);
    }
    for (int k0 = 0; k0 < HIDW; k0 += 32) {
        v16bf a = load_a_frag(agghbf, HIDW, node_base, k0, lane);
        accz = WMMA_BF16(a, load_b_frag(Wlhz, HIDW, col_base, k0, lane), accz);
        accr = WMMA_BF16(a, load_b_frag(Wlhr, HIDW, col_base, k0, lane), accr);
        a = load_a_frag(hbf, HIDW, node_base, k0, lane);
        accz = WMMA_BF16(a, load_b_frag(Wrhz, HIDW, col_base, k0, lane), accz);
        accr = WMMA_BF16(a, load_b_frag(Wrhr, HIDW, col_base, k0, lane), accr);
    }

    int col = col_base + (lane & 15);
    int kg = lane >> 4;
    float bz = bxz[col] + bhz[col];
    float br = bxr[col] + bhr[col];
#pragma unroll
    for (int v = 0; v < 8; ++v) {
        int node = node_base + v + 8 * kg;
        size_t o = (size_t)node * HIDW + col;
        float z = 1.0f / (1.0f + __expf(-(accz[v] + bz)));
        float r = 1.0f / (1.0f + __expf(-(accr[v] + br)));
        float h = hfull[o];
        zout[o] = z;
        rhout[o] = r * h;
    }
}

// ---------------- fused h_tilde GEMM + GRU output ----------------
// ht = tanh(agg_x@Wlxh^T + x@Wrxh^T + agg_rh@Wlhh^T + rh@Wrhh^T + bxh + bhh)
// out = z*h + (1-z)*ht ; also emits bf16 copy for next layer's input
__global__ void __launch_bounds__(256)
htilde_kernel(const __bf16* __restrict__ aggx, const __bf16* __restrict__ xbf, int Kx,
              const __bf16* __restrict__ aggrhbf, const __bf16* __restrict__ rhbf,
              const __bf16* __restrict__ Wlxh, const __bf16* __restrict__ Wrxh,
              const __bf16* __restrict__ Wlhh, const __bf16* __restrict__ Wrhh,
              const float* __restrict__ bxh, const float* __restrict__ bhh,
              const float* __restrict__ zbuf, const float* __restrict__ hfull,
              float* __restrict__ out, __bf16* __restrict__ outbf) {
    int lane = threadIdx.x & 31;
    int wave = threadIdx.x >> 5;
    int node_base = blockIdx.x * 16;
    int col_base = wave * 16;

    v8f acc = {};
    for (int k0 = 0; k0 < Kx; k0 += 32) {
        v16bf a = load_a_frag(aggx, Kx, node_base, k0, lane);
        acc = WMMA_BF16(a, load_b_frag(Wlxh, Kx, col_base, k0, lane), acc);
        a = load_a_frag(xbf, Kx, node_base, k0, lane);
        acc = WMMA_BF16(a, load_b_frag(Wrxh, Kx, col_base, k0, lane), acc);
    }
    for (int k0 = 0; k0 < HIDW; k0 += 32) {
        v16bf a = load_a_frag(aggrhbf, HIDW, node_base, k0, lane);
        acc = WMMA_BF16(a, load_b_frag(Wlhh, HIDW, col_base, k0, lane), acc);
        a = load_a_frag(rhbf, HIDW, node_base, k0, lane);
        acc = WMMA_BF16(a, load_b_frag(Wrhh, HIDW, col_base, k0, lane), acc);
    }

    int col = col_base + (lane & 15);
    int kg = lane >> 4;
    float b = bxh[col] + bhh[col];
#pragma unroll
    for (int v = 0; v < 8; ++v) {
        int node = node_base + v + 8 * kg;
        size_t o = (size_t)node * HIDW + col;
        float ht = tanhf(acc[v] + b);
        float z = zbuf[o];
        float res = z * hfull[o] + (1.0f - z) * ht;
        out[o] = res;
        outbf[o] = (__bf16)res;
    }
}

// ---------------- host-side orchestration ----------------
static inline int cdiv(int a, int b) { return (a + b - 1) / b; }

extern "C" void kernel_launch(void* const* d_in, const int* in_sizes, int n_in,
                              void* d_out, int out_size, void* d_ws, size_t ws_size,
                              hipStream_t stream) {
    (void)in_sizes; (void)n_in; (void)out_size; (void)ws_size;
    const float* x   = (const float*)d_in[0];
    const int*   ei  = (const int*)d_in[1];
    const int*   src = ei;
    const int*   dst = ei + EE;
    const float* hin = (const float*)d_in[2];
    float* out = (float*)d_out;

    // params flattened in JAX sorted-dict order: gates hh,hr,hz,xh,xr,xz ; each (Wl,Wr,b)
    auto P = [&](int l, int g, int c) -> const float* {
        return (const float*)d_in[3 + l * 18 + g * 3 + c];
    };

    // workspace carve-up (256B aligned)
    char* ws = (char*)d_ws;
    size_t off = 0;
    auto alloc = [&](size_t bytes) -> void* {
        void* p = ws + off;
        off = (off + bytes + 255) & ~(size_t)255;
        return p;
    };
    float*  deg     = (float*) alloc((size_t)NN * 4);
    float*  inv     = (float*) alloc((size_t)NN * 4);
    float*  aggx    = (float*) alloc((size_t)NN * DIN * 4);
    float*  aggh    = (float*) alloc((size_t)NN * HIDW * 4);
    float*  aggrh   = (float*) alloc((size_t)NN * HIDW * 4);
    float*  rh      = (float*) alloc((size_t)NN * HIDW * 4);
    float*  zbuf    = (float*) alloc((size_t)NN * HIDW * 4);
    __bf16* xbf     = (__bf16*)alloc((size_t)NN * DIN * 2);
    __bf16* aggxbf  = (__bf16*)alloc((size_t)NN * DIN * 2);
    __bf16* hbf     = (__bf16*)alloc((size_t)NN * HIDW * 2);
    __bf16* agghbf  = (__bf16*)alloc((size_t)NN * HIDW * 2);
    __bf16* rhbf    = (__bf16*)alloc((size_t)NN * HIDW * 2);
    __bf16* aggrhbf = (__bf16*)alloc((size_t)NN * HIDW * 2);
    __bf16* outbf0  = (__bf16*)alloc((size_t)NN * HIDW * 2);
    __bf16* outbf1  = (__bf16*)alloc((size_t)NN * HIDW * 2);
    __bf16* wpool   = (__bf16*)alloc((size_t)(6 * 128 * 256 + 6 * 128 * 128 + 12 * 128 * 128) * 2);

    // ---- convert all weight matrices to bf16 once ----
    __bf16* wbf[NLAYERS][6][2];
    size_t woff = 0;
    for (int l = 0; l < NLAYERS; ++l)
        for (int g = 0; g < 6; ++g)
            for (int c = 0; c < 2; ++c) {
                int din = (g >= 3) ? ((l == 0) ? DIN : HIDW) : HIDW;
                int n = HIDW * din;
                wbf[l][g][c] = wpool + woff;
                woff += (size_t)n;
                cvt_bf16_kernel<<<cdiv(n, 256), 256, 0, stream>>>(P(l, g, c), wbf[l][g][c], n);
            }

    // ---- degrees (shared across layers) ----
    fill_zero_kernel<<<cdiv(NN, 256), 256, 0, stream>>>(deg, NN);
    deg_count_kernel<<<cdiv(EE, 256), 256, 0, stream>>>(dst, deg, EE);
    inv_deg_kernel<<<cdiv(NN, 256), 256, 0, stream>>>(deg, inv, NN);

    const int tiles = NN / 16;  // 1250

    for (int l = 0; l < NLAYERS; ++l) {
        const int Kx = (l == 0) ? DIN : HIDW;
        const int dsh = (Kx == DIN) ? 8 : 7;
        const float*  inp   = (l == 0) ? x : out;      // layer-1 input = layer-0 output
        const float*  hi    = hin + (size_t)l * NN * HIDW;
        float*        outl  = out + (size_t)l * NN * HIDW;
        __bf16*       obf   = (l == 0) ? outbf0 : outbf1;
        const __bf16* inpbf;
        if (l == 0) {
            cvt_bf16_kernel<<<cdiv(NN * DIN, 256), 256, 0, stream>>>(x, xbf, NN * DIN);
            inpbf = xbf;
        } else {
            inpbf = outbf0;
        }

        // agg_x = mean_{j->i} inp_j
        fill_zero_kernel<<<cdiv(NN * Kx, 256), 256, 0, stream>>>(aggx, NN * Kx);
        {
            int total = EE << (dsh - 2);
            scatter_add_kernel<<<cdiv(total, 256), 256, 0, stream>>>(inp, src, dst, aggx, dsh, total);
        }
        scale_cvt_kernel<<<cdiv(NN * Kx, 256), 256, 0, stream>>>(aggx, inv, aggxbf, dsh, NN * Kx);

        // h and agg_h
        cvt_bf16_kernel<<<cdiv(NN * HIDW, 256), 256, 0, stream>>>(hi, hbf, NN * HIDW);
        fill_zero_kernel<<<cdiv(NN * HIDW, 256), 256, 0, stream>>>(aggh, NN * HIDW);
        {
            int total = EE << 5;  // 128/4 = 32 chunks per edge
            scatter_add_kernel<<<cdiv(total, 256), 256, 0, stream>>>(hi, src, dst, aggh, 7, total);
        }
        scale_cvt_kernel<<<cdiv(NN * HIDW, 256), 256, 0, stream>>>(aggh, inv, agghbf, 7, NN * HIDW);

        // z and r gates (fused), emits z and rh = r*h
        gate_zr_kernel<<<tiles, 256, 0, stream>>>(
            aggxbf, inpbf, Kx, agghbf, hbf,
            wbf[l][5][0], wbf[l][5][1], wbf[l][2][0], wbf[l][2][1],   // z: xz.Wl, xz.Wr, hz.Wl, hz.Wr
            wbf[l][4][0], wbf[l][4][1], wbf[l][1][0], wbf[l][1][1],   // r: xr.Wl, xr.Wr, hr.Wl, hr.Wr
            P(l, 5, 2), P(l, 2, 2), P(l, 4, 2), P(l, 1, 2),           // biases: xz, hz, xr, hr
            hi, zbuf, rh);

        // agg(rh)
        cvt_bf16_kernel<<<cdiv(NN * HIDW, 256), 256, 0, stream>>>(rh, rhbf, NN * HIDW);
        fill_zero_kernel<<<cdiv(NN * HIDW, 256), 256, 0, stream>>>(aggrh, NN * HIDW);
        {
            int total = EE << 5;
            scatter_add_kernel<<<cdiv(total, 256), 256, 0, stream>>>(rh, src, dst, aggrh, 7, total);
        }
        scale_cvt_kernel<<<cdiv(NN * HIDW, 256), 256, 0, stream>>>(aggrh, inv, aggrhbf, 7, NN * HIDW);

        // h_tilde + GRU combine
        htilde_kernel<<<tiles, 256, 0, stream>>>(
            aggxbf, inpbf, Kx, aggrhbf, rhbf,
            wbf[l][3][0], wbf[l][3][1], wbf[l][0][0], wbf[l][0][1],   // xh.Wl, xh.Wr, hh.Wl, hh.Wr
            P(l, 3, 2), P(l, 0, 2),                                   // biases: xh, hh
            zbuf, hi, outl, obf);
    }
}
